// MixtureOfExperts_19825569038698
// MI455X (gfx1250) — matrix operand order repaired and checked by reading, not compiled, not run on hardware
//
#include <hip/hip_runtime.h>
#include <hip/hip_bf16.h>

// ---------------- problem constants ----------------
#define B_  16384
#define D_  256
#define H_  512
#define O_  256
#define E_  16
#define K_  2
#define TM  32                // tokens per expert tile
#define NSLOT (B_ * K_)       // 32768 routing slots

typedef __bf16 bf16_t;
typedef __bf16  v16bf __attribute__((ext_vector_type(16)));
typedef float   v8f   __attribute__((ext_vector_type(8)));

union Frag {
    v16bf v;
    uint4 u[2];
};

#define WMMA_BF16(A, Bf, C) \
    __builtin_amdgcn_wmma_f32_16x16x32_bf16(false, (A), false, (Bf), (short)0, (C), false, false)

// ---------------- kernel 1: zero output + counters ----------------
__global__ __launch_bounds__(256) void moe_zero_kernel(float* out, int* counts) {
    int i = blockIdx.x * 256 + threadIdx.x;
    if (i < B_ * O_) out[i] = 0.0f;
    if (i < E_) counts[i] = 0;
}

// ---------------- kernel 2: gating (softmax + top-2) ----------------
__global__ __launch_bounds__(256) void moe_gate_kernel(
    const float* __restrict__ x, const float* __restrict__ wg,
    int* __restrict__ topi, float* __restrict__ topg, int* __restrict__ counts)
{
    __shared__ float wgs[D_ * E_];                 // 16 KB
    for (int i = threadIdx.x; i < D_ * E_; i += 256) wgs[i] = wg[i];
    __syncthreads();

    int b = blockIdx.x * 256 + threadIdx.x;
    if (b >= B_) return;
    const float* xr = x + (size_t)b * D_;

    float acc[E_];
    #pragma unroll
    for (int e = 0; e < E_; ++e) acc[e] = 0.0f;
    for (int d = 0; d < D_; d += 4) {
        float4 xv = *(const float4*)(xr + d);
        #pragma unroll
        for (int e = 0; e < E_; ++e) {
            acc[e] += xv.x * wgs[(d + 0) * E_ + e];
            acc[e] += xv.y * wgs[(d + 1) * E_ + e];
            acc[e] += xv.z * wgs[(d + 2) * E_ + e];
            acc[e] += xv.w * wgs[(d + 3) * E_ + e];
        }
    }
    // softmax over E
    float mx = acc[0];
    #pragma unroll
    for (int e = 1; e < E_; ++e) mx = fmaxf(mx, acc[e]);
    float s = 0.0f;
    #pragma unroll
    for (int e = 0; e < E_; ++e) { acc[e] = expf(acc[e] - mx); s += acc[e]; }
    float inv = 1.0f / s;
    #pragma unroll
    for (int e = 0; e < E_; ++e) acc[e] *= inv;
    // top-2 (strict > keeps lowest index on ties, matching lax.top_k)
    int   i0 = 0;  float g0 = acc[0];
    #pragma unroll
    for (int e = 1; e < E_; ++e) if (acc[e] > g0) { g0 = acc[e]; i0 = e; }
    int   i1 = -1; float g1 = -1.0f;
    #pragma unroll
    for (int e = 0; e < E_; ++e) if (e != i0 && acc[e] > g1) { g1 = acc[e]; i1 = e; }

    topi[b * 2 + 0] = i0; topg[b * 2 + 0] = g0;
    topi[b * 2 + 1] = i1; topg[b * 2 + 1] = g1;
    atomicAdd(&counts[i0], 1);
    atomicAdd(&counts[i1], 1);
}

// ---------------- kernel 3: exclusive scan over 16 experts ----------------
__global__ void moe_scan_kernel(const int* counts, int* offsets, int* cursor) {
    if (blockIdx.x == 0 && threadIdx.x == 0) {
        int acc = 0;
        for (int e = 0; e < E_; ++e) { offsets[e] = acc; cursor[e] = acc; acc += counts[e]; }
    }
}

// ---------------- kernel 4: scatter slots into expert buckets ----------------
__global__ __launch_bounds__(256) void moe_scatter_kernel(
    const int* __restrict__ topi, const float* __restrict__ topg,
    int* __restrict__ cursor, int* __restrict__ stok, float* __restrict__ sgate)
{
    int s = blockIdx.x * 256 + threadIdx.x;
    if (s >= NSLOT) return;
    int e = topi[s];
    int p = atomicAdd(&cursor[e], 1);
    stok[p]  = s / K_;
    sgate[p] = topg[s];
}

// ---------------- kernel 5a/5b: transpose+convert weights to bf16 ----------------
__global__ __launch_bounds__(256) void moe_t_w1_kernel(const float* __restrict__ W1, bf16_t* __restrict__ w1t) {
    int i = blockIdx.x * 256 + threadIdx.x;           // over [E][H][D]
    if (i >= E_ * H_ * D_) return;
    int d = i % D_; int h = (i / D_) % H_; int e = i / (D_ * H_);
    w1t[i] = (bf16_t)W1[((size_t)e * D_ + d) * H_ + h];
}
__global__ __launch_bounds__(256) void moe_t_w2_kernel(const float* __restrict__ W2, bf16_t* __restrict__ w2t) {
    int i = blockIdx.x * 256 + threadIdx.x;           // over [E][O][H]
    if (i >= E_ * O_ * H_) return;
    int h = i % H_; int o = (i / H_) % O_; int e = i / (H_ * O_);
    w2t[i] = (bf16_t)W2[((size_t)e * H_ + h) * O_ + o];
}

// ---------------- kernel 6: per-expert fused MLP (WMMA bf16) ----------------
// block = 256 threads (8 waves); tile = TM(32) tokens of one expert bucket.
// Each wave computes BOTH 16-row M-tiles against its private N columns so every
// B fragment (global b128 pair) feeds two independent WMMAs.
__global__ __launch_bounds__(256) void moe_expert_kernel(
    const float*  __restrict__ x,       // [B, D]
    const bf16_t* __restrict__ w1t,     // [E, H, D]
    const float*  __restrict__ b1,      // [E, H]
    const bf16_t* __restrict__ w2t,     // [E, O, H]
    const float*  __restrict__ b2,      // [E, O]
    const int*    __restrict__ stok,
    const float*  __restrict__ sgate,
    const int*    __restrict__ offsets,
    const int*    __restrict__ counts,
    float*        __restrict__ out)     // [B, O]
{
    __shared__ bf16_t xs[TM][D_];       // 16 KB
    __shared__ bf16_t hs[TM][H_];       // 32 KB
    __shared__ int    tok_s[TM];
    __shared__ float  gate_s[TM];

    const int e   = blockIdx.y;
    const int cnt = counts[e];
    if ((int)blockIdx.x * TM >= cnt) return;
    const int base  = offsets[e] + blockIdx.x * TM;
    const int nrows = min(TM, cnt - (int)blockIdx.x * TM);

    const int tid = threadIdx.x;
    if (tid < TM) {
        if (tid < nrows) { tok_s[tid] = stok[base + tid]; gate_s[tid] = sgate[base + tid]; }
        else             { tok_s[tid] = 0;                gate_s[tid] = 0.0f; }
    }
    __syncthreads();

    // gather x rows -> LDS as bf16 (zero the pad rows)
    for (int i = tid; i < TM * (D_ / 4); i += 256) {
        int r  = i / (D_ / 4);
        int c4 = i % (D_ / 4);
        float4 v = make_float4(0.f, 0.f, 0.f, 0.f);
        if (r < nrows) v = ((const float4*)(x + (size_t)tok_s[r] * D_))[c4];
        bf16_t* dst = &xs[r][c4 * 4];
        dst[0] = (bf16_t)v.x; dst[1] = (bf16_t)v.y; dst[2] = (bf16_t)v.z; dst[3] = (bf16_t)v.w;
    }
    __syncthreads();

    const int lane   = tid & 31;
    const int wv     = tid >> 5;          // 0..7
    const int lrow   = lane & 15;
    const int khalfA = (lane >> 4) * 8;   // A frag: K in {kh..kh+7, kh+16..kh+23}
    const int khalfB = (lane >> 4) * 16;  // unused: B frag loads use khalfA, same K order as A
    const int moff   = (lane >> 4) * 8;   // C/D: VGPR i -> row i (+8 for hi lanes)

    // ---- GEMM1: h = relu(x @ W1 + b1), C tile [32, 512] ----
    // wave wv owns hidden columns [wv*64, wv*64+64): 4 N-tiles x 2 M-tiles.
    {
        int ncol[4];
        const bf16_t* brow[4];
        #pragma unroll
        for (int j = 0; j < 4; ++j) {
            ncol[j] = (wv * 4 + j) * 16 + lrow;
            brow[j] = w1t + ((size_t)e * H_ + ncol[j]) * D_;
        }
        v8f acc[2][4];
        #pragma unroll
        for (int mt = 0; mt < 2; ++mt)
            #pragma unroll
            for (int j = 0; j < 4; ++j)
                acc[mt][j] = (v8f){0.f, 0.f, 0.f, 0.f, 0.f, 0.f, 0.f, 0.f};

        #pragma unroll
        for (int ks = 0; ks < 8; ++ks) {
            const int kbA = ks * 32 + khalfA;
            const int kbB = ks * 32 + khalfA;
            Frag a0, a1;
            a0.u[0] = *(const uint4*)&xs[lrow][kbA];
            a0.u[1] = *(const uint4*)&xs[lrow][kbA + 16];
            a1.u[0] = *(const uint4*)&xs[16 + lrow][kbA];
            a1.u[1] = *(const uint4*)&xs[16 + lrow][kbA + 16];
            #pragma unroll
            for (int j = 0; j < 4; ++j) {
                Frag bf;
                bf.u[0] = *(const uint4*)(brow[j] + kbB);
                bf.u[1] = *(const uint4*)(brow[j] + kbB + 16);
                acc[0][j] = WMMA_BF16(a0.v, bf.v, acc[0][j]);
                acc[1][j] = WMMA_BF16(a1.v, bf.v, acc[1][j]);
            }
        }
        #pragma unroll
        for (int j = 0; j < 4; ++j) {
            const float bias = b1[e * H_ + ncol[j]];
            #pragma unroll
            for (int mt = 0; mt < 2; ++mt) {
                #pragma unroll
                for (int i = 0; i < 8; ++i) {
                    const int m = mt * 16 + i + moff;
                    const float hv = acc[mt][j][i] + bias;
                    hs[m][ncol[j]] = (bf16_t)(hv > 0.0f ? hv : 0.0f);
                }
            }
        }
    }
    __syncthreads();

    // ---- GEMM2: out_slot = h @ W2 + b2, C tile [32, 256] ----
    // wave wv owns output columns [wv*32, wv*32+32): 2 N-tiles x 2 M-tiles.
    {
        int ocol[2];
        const bf16_t* brow[2];
        #pragma unroll
        for (int j = 0; j < 2; ++j) {
            ocol[j] = (wv * 2 + j) * 16 + lrow;
            brow[j] = w2t + ((size_t)e * O_ + ocol[j]) * H_;
        }
        v8f acc[2][2];
        #pragma unroll
        for (int mt = 0; mt < 2; ++mt)
            #pragma unroll
            for (int j = 0; j < 2; ++j)
                acc[mt][j] = (v8f){0.f, 0.f, 0.f, 0.f, 0.f, 0.f, 0.f, 0.f};

        #pragma unroll
        for (int ks = 0; ks < 16; ++ks) {
            const int kbA = ks * 32 + khalfA;
            const int kbB = ks * 32 + khalfA;
            Frag a0, a1;
            a0.u[0] = *(const uint4*)&hs[lrow][kbA];
            a0.u[1] = *(const uint4*)&hs[lrow][kbA + 16];
            a1.u[0] = *(const uint4*)&hs[16 + lrow][kbA];
            a1.u[1] = *(const uint4*)&hs[16 + lrow][kbA + 16];
            #pragma unroll
            for (int j = 0; j < 2; ++j) {
                Frag bf;
                bf.u[0] = *(const uint4*)(brow[j] + kbB);
                bf.u[1] = *(const uint4*)(brow[j] + kbB + 16);
                acc[0][j] = WMMA_BF16(a0.v, bf.v, acc[0][j]);
                acc[1][j] = WMMA_BF16(a1.v, bf.v, acc[1][j]);
            }
        }

        float grow[2][8]; int trow[2][8]; int mvalid[2][8];
        #pragma unroll
        for (int mt = 0; mt < 2; ++mt) {
            #pragma unroll
            for (int i = 0; i < 8; ++i) {
                const int m = mt * 16 + i + moff;
                grow[mt][i] = gate_s[m]; trow[mt][i] = tok_s[m]; mvalid[mt][i] = (m < nrows);
            }
        }
        #pragma unroll
        for (int j = 0; j < 2; ++j) {
            const float bias = b2[e * O_ + ocol[j]];
            #pragma unroll
            for (int mt = 0; mt < 2; ++mt) {
                #pragma unroll
                for (int i = 0; i < 8; ++i) {
                    if (mvalid[mt][i]) {
                        const float v = (acc[mt][j][i] + bias) * grow[mt][i];
                        atomicAdd(out + (size_t)trow[mt][i] * O_ + ocol[j], v);
                    }
                }
            }
        }
    }
}

// ---------------- launcher ----------------
extern "C" void kernel_launch(void* const* d_in, const int* in_sizes, int n_in,
                              void* d_out, int out_size, void* d_ws, size_t ws_size,
                              hipStream_t stream) {
    const float* x  = (const float*)d_in[0];   // [B, D]
    const float* wg = (const float*)d_in[1];   // [D, E]
    const float* W1 = (const float*)d_in[2];   // [E, D, H]
    const float* b1 = (const float*)d_in[3];   // [E, H]
    const float* W2 = (const float*)d_in[4];   // [E, H, O]
    const float* b2 = (const float*)d_in[5];   // [E, O]
    float* out = (float*)d_out;                // [B, O]

    // workspace carve-out (256B aligned segments)
    size_t off = 0;
    auto carve = [&](size_t bytes) -> void* {
        void* p = (char*)d_ws + off;
        off += (bytes + 255) & ~(size_t)255;
        return p;
    };
    int*    topi    = (int*)   carve((size_t)NSLOT * 4);
    float*  topg    = (float*) carve((size_t)NSLOT * 4);
    int*    counts  = (int*)   carve(E_ * 4);
    int*    offsets = (int*)   carve(E_ * 4);
    int*    cursor  = (int*)   carve(E_ * 4);
    int*    stok    = (int*)   carve((size_t)NSLOT * 4);
    float*  sgate   = (float*) carve((size_t)NSLOT * 4);
    bf16_t* w1t     = (bf16_t*)carve((size_t)E_ * H_ * D_ * 2);
    bf16_t* w2t     = (bf16_t*)carve((size_t)E_ * O_ * H_ * 2);

    moe_zero_kernel<<<(B_ * O_ + 255) / 256, 256, 0, stream>>>(out, counts);
    moe_gate_kernel<<<B_ / 256, 256, 0, stream>>>(x, wg, topi, topg, counts);
    moe_scan_kernel<<<1, 1, 0, stream>>>(counts, offsets, cursor);
    moe_scatter_kernel<<<(NSLOT + 255) / 256, 256, 0, stream>>>(topi, topg, cursor, stok, sgate);
    moe_t_w1_kernel<<<(E_ * H_ * D_ + 255) / 256, 256, 0, stream>>>(W1, w1t);
    moe_t_w2_kernel<<<(E_ * O_ * H_ + 255) / 256, 256, 0, stream>>>(W2, w2t);

    dim3 grid(NSLOT / TM, E_);   // worst case: one expert owns every slot
    moe_expert_kernel<<<grid, 256, 0, stream>>>(x, w1t, b1, w2t, b2,
                                                stok, sgate, offsets, counts, out);
}
